// EncoderLayer_11278584119371
// MI455X (gfx1250) — matrix operand order repaired and checked by reading, not compiled, not run on hardware
//
#include <hip/hip_runtime.h>
#include <hip/hip_bf16.h>
#include <math.h>
#include <stdint.h>

// ---------------- problem constants ----------------
constexpr int Bn   = 2;
constexpr int Tn   = 2048;
constexpr int Dn   = 256;    // per-head dim AND model dim
constexpr int Hn   = 8;
constexpr int DFFn = 1024;
constexpr int HD   = Hn * Dn;      // 2048
constexpr int Mrows = Bn * Tn;     // 4096

typedef _Float16 half_t;
typedef __attribute__((ext_vector_type(16))) _Float16 v16h;
typedef __attribute__((ext_vector_type(8)))  _Float16 v8h;
typedef __attribute__((ext_vector_type(8)))  float    v8f;

// ---------------- fragment loaders ----------------
// A-matrix 16x32 f16 fragment (contiguous k order used here, not the ISA table order): lane l holds row m = l%16.
// elements e=0..15 -> k = base+e, base = (l&16)?16:0.
__device__ __forceinline__ v16h load_a_frag(const half_t* __restrict__ A,
                                            int row, int ldk, int kb, int lane) {
    const half_t* p = A + (size_t)row * ldk + kb * 32 + ((lane & 16) ? 16 : 0);
    v8h lo = *(const v8h*)p;          // 16B aligned
    v8h hi = *(const v8h*)(p + 8);    // 16B aligned
    v16h r;
#pragma unroll
    for (int e = 0; e < 8; ++e) { r[e] = lo[e]; r[8 + e] = hi[e]; }
    return r;
}

// B-matrix 32x16 f16 fragment: lane l holds column n = l%16,
// k = (l<16 ? 0 : 16) + e, e=0..15 (contiguous 16 halfs = one v16h load).

// ---------------- weight repack: fp32 [K,N] -> packed f16 B-fragments ----------
__global__ void pack_w_kernel(const float* __restrict__ W, half_t* __restrict__ Wp,
                              int K, int N) {
    int lane = threadIdx.x;           // 32
    int kb = blockIdx.x;              // K/32
    int nb = blockIdx.y;              // N/16
    int KB = K / 32;
    int n = nb * 16 + (lane & 15);
    int kbase = kb * 32 + ((lane & 16) ? 16 : 0);
    half_t* dst = Wp + (((size_t)nb * KB + kb) * 32 + lane) * 16;
#pragma unroll
    for (int e = 0; e < 16; ++e)
        dst[e] = (half_t)W[(size_t)(kbase + e) * N + n];
}

// ---------------- LayerNorm (fp32 in) -> f16 out, one row per block ----------
__global__ void layernorm_f16_kernel(const float* __restrict__ x,
                                     const float* __restrict__ w,
                                     const float* __restrict__ b,
                                     half_t* __restrict__ out) {
    __shared__ float red[Dn];
    size_t row = blockIdx.x;
    int t = threadIdx.x;                       // Dn == 256 threads
    float v = x[row * Dn + t];
    red[t] = v; __syncthreads();
    for (int s = Dn / 2; s > 0; s >>= 1) {
        if (t < s) red[t] += red[t + s];
        __syncthreads();
    }
    float mean = red[0] / (float)Dn;
    __syncthreads();
    float dv = v - mean;
    red[t] = dv * dv; __syncthreads();
    for (int s = Dn / 2; s > 0; s >>= 1) {
        if (t < s) red[t] += red[t + s];
        __syncthreads();
    }
    float var = red[0] / (float)Dn;
    float r = rsqrtf(var + 1e-6f);
    out[row * Dn + t] = (half_t)(dv * r * w[t] + b[t]);
}

// ---------------- generic WMMA GEMM, one wave -> 16x64 output tile -----------
enum { OUT_QK = 0, OUT_VT = 1, OUT_F16RELU = 2, OUT_F32RES = 3 };

template <int MODE>
__global__ void gemm_wmma_kernel(const half_t* __restrict__ A,
                                 const half_t* __restrict__ Bp,
                                 const float* __restrict__ bias,
                                 const float* __restrict__ res,
                                 float* __restrict__ outF,
                                 half_t* __restrict__ outH,
                                 int N, int K) {
    int lane = threadIdx.x;
    int nb64 = blockIdx.x;           // N/64 tile groups
    int mb = blockIdx.y;
    int KB = K / 32;
    v8f acc[4] = {{}, {}, {}, {}};
    const half_t* bbase[4];
#pragma unroll
    for (int s = 0; s < 4; ++s)
        bbase[s] = Bp + (size_t)(nb64 * 4 + s) * KB * 512;   // 32 lanes * 16 halfs
    int arow = mb * 16 + (lane & 15);
    const half_t* aptr = A + (size_t)arow * K + ((lane & 16) ? 8 : 0);
    for (int kb = 0; kb < KB; ++kb) {
        if (kb + 1 < KB)
            __builtin_prefetch(aptr + (kb + 1) * 32, 0, 3);  // global_prefetch_b8
        v16h a = load_a_frag(A, arow, K, kb, lane);
#pragma unroll
        for (int s = 0; s < 4; ++s) {
            v16h b = *(const v16h*)(bbase[s] + ((size_t)kb * 32 + lane) * 16);
            acc[s] = __builtin_amdgcn_wmma_f32_16x16x32_f16(false, a, false, b,
                                                            (short)0, acc[s], false, false);
        }
    }
#pragma unroll
    for (int s = 0; s < 4; ++s) {
        int n = (nb64 * 4 + s) * 16 + (lane & 15);
        float bv = bias ? bias[n] : 0.0f;
#pragma unroll
        for (int r = 0; r < 8; ++r) {
            int m = mb * 16 + r + ((lane & 16) ? 8 : 0);
            float val = acc[s][r] + bv;
            if (MODE == OUT_F32RES) {
                size_t idx = (size_t)m * N + n;
                outF[idx] = val + (res ? res[idx] : 0.0f);
            } else if (MODE == OUT_F16RELU) {
                outH[(size_t)m * N + n] = (half_t)(val > 0.0f ? val : 0.0f);
            } else if (MODE == OUT_QK) {
                int bb = m / Tn, t = m % Tn, h = n / Dn, d = n % Dn;
                outH[(((size_t)(bb * Hn + h) * Tn + t) * Dn) + d] = (half_t)val;
            } else { // OUT_VT: transposed value cache [B,H,D,T]
                int bb = m / Tn, t = m % Tn, h = n / Dn, d = n % Dn;
                outH[(((size_t)(bb * Hn + h) * Dn + d) * Tn) + t] = (half_t)val;
            }
        }
    }
}

// ---------------- attention scores: S = Q K^T / 16, one wave -> 16x64 --------
__global__ void attn_scores_kernel(const half_t* __restrict__ q,
                                   const half_t* __restrict__ k,
                                   const int* __restrict__ mask,
                                   float* __restrict__ attn) {
    int lane = threadIdx.x;
    int jb64 = blockIdx.x;           // Tn/64 key tile groups
    int ib = blockIdx.y;             // Tn/16 query tiles
    int bh = blockIdx.z;             // B*H
    const half_t* Q  = q + (size_t)bh * Tn * Dn;
    const half_t* Km = k + (size_t)bh * Tn * Dn;
    v8f acc[4] = {{}, {}, {}, {}};
    int arow = ib * 16 + (lane & 15);
    for (int kb = 0; kb < Dn / 32; ++kb) {
        v16h a = load_a_frag(Q, arow, Dn, kb, lane);
#pragma unroll
        for (int s = 0; s < 4; ++s) {
            int brow = jb64 * 64 + s * 16 + (lane & 15);
            v16h b = *(const v16h*)(Km + (size_t)brow * Dn + kb * 32 + ((lane & 16) ? 16 : 0));
            acc[s] = __builtin_amdgcn_wmma_f32_16x16x32_f16(false, a, false, b,
                                                            (short)0, acc[s], false, false);
        }
    }
    const float scale = 0.0625f;     // 1/sqrt(256)
#pragma unroll
    for (int s = 0; s < 4; ++s) {
        int j = jb64 * 64 + s * 16 + (lane & 15);
#pragma unroll
        for (int r = 0; r < 8; ++r) {
            int i = ib * 16 + r + ((lane & 16) ? 8 : 0);
            float v = acc[s][r] * scale;
            if (mask[(size_t)i * Tn + j] == 0) v = -INFINITY;
            attn[((size_t)bh * Tn + i) * Tn + j] = v;
        }
    }
}

// ---------------- softmax over last dim, in place ----------------------------
__global__ void softmax_rows_kernel(float* __restrict__ attn) {
    __shared__ float red[256];
    size_t row = blockIdx.x;
    float* p = attn + row * (size_t)Tn;
    int t = threadIdx.x;
    float m = -INFINITY;
    for (int c = t; c < Tn; c += 256) m = fmaxf(m, p[c]);
    red[t] = m; __syncthreads();
    for (int s = 128; s > 0; s >>= 1) {
        if (t < s) red[t] = fmaxf(red[t], red[t + s]);
        __syncthreads();
    }
    m = red[0];
    __syncthreads();
    float sum = 0.0f;
    for (int c = t; c < Tn; c += 256) {
        float e = __expf(p[c] - m);
        p[c] = e;
        sum += e;
    }
    red[t] = sum; __syncthreads();
    for (int s = 128; s > 0; s >>= 1) {
        if (t < s) red[t] += red[t + s];
        __syncthreads();
    }
    float inv = 1.0f / red[0];
    for (int c = t; c < Tn; c += 256) p[c] *= inv;
}

// ---------------- out = attn @ V, cooperative block (4 waves, full D) --------
// Block = 128 threads; the fp32 attn A-tile (16x32) is staged to LDS ONCE per
// K-step via CDNA5 async global->LDS copy, shared by all 4 waves; each wave
// produces a 16x64 slice of the 16x256 output row-tile. attn is read from HBM
// exactly once across the whole kernel.
__global__ void attn_out_kernel(const float* __restrict__ attn,
                                const half_t* __restrict__ vt,
                                half_t* __restrict__ ao) {
    __shared__ float atile[16 * 32];
    int tid  = threadIdx.x;          // 128
    int lane = tid & 31;
    int wave = tid >> 5;             // 0..3 -> 64-wide d slice
    int ib = blockIdx.x;             // Tn/16 query tiles
    int bh = blockIdx.y;             // B*H
    const float*  Am = attn + (size_t)bh * Tn * Tn;
    const half_t* Vt = vt   + (size_t)bh * Dn * Tn;
    v8f acc[4] = {{}, {}, {}, {}};

    int rrow = tid >> 3;             // 0..15: staged row
    int rc4  = (tid & 7) * 4;        // 0,4,..28: staged float4 column
    uint32_t ldsdst = (uint32_t)(uintptr_t)&atile[rrow * 32 + rc4];  // low 32b = LDS offset
    const float* gsrc = Am + (size_t)(ib * 16 + rrow) * Tn + rc4;
    const float* lrow = &atile[(lane & 15) * 32 + ((lane & 16) ? 16 : 0)];
    int bcol = wave * 64 + (lane & 15);

    for (int kb = 0; kb < Tn / 32; ++kb) {
        // async copy 16B/thread: 128 threads cover the 2KB fp32 A tile
        uint64_t ga = (uint64_t)(uintptr_t)(gsrc + kb * 32);
        asm volatile("global_load_async_to_lds_b128 %0, %1, off"
                     :: "v"(ldsdst), "v"(ga) : "memory");
        asm volatile("s_wait_asynccnt 0x0" ::: "memory");
        __syncthreads();
        // build f16 A fragment from the shared LDS tile
        v16h a;
#pragma unroll
        for (int e = 0; e < 8; ++e) {
            a[e]     = (half_t)lrow[e];
            a[8 + e] = (half_t)lrow[8 + e];
        }
#pragma unroll
        for (int s = 0; s < 4; ++s) {
            v16h b = *(const v16h*)(Vt + (size_t)(bcol + s * 16) * Tn + kb * 32
                                    + ((lane & 16) ? 16 : 0));
            acc[s] = __builtin_amdgcn_wmma_f32_16x16x32_f16(false, a, false, b,
                                                            (short)0, acc[s], false, false);
        }
        __syncthreads();             // protect LDS tile before next overwrite
    }
    int h = bh % Hn, bb = bh / Hn;
#pragma unroll
    for (int s = 0; s < 4; ++s) {
        int d = wave * 64 + s * 16 + (lane & 15);
#pragma unroll
        for (int r = 0; r < 8; ++r) {
            int i = ib * 16 + r + ((lane & 16) ? 8 : 0);
            ao[((size_t)(bb * Tn + i)) * HD + h * Dn + d] = (half_t)acc[s][r];
        }
    }
}

// ---------------- host-side orchestration ------------------------------------
extern "C" void kernel_launch(void* const* d_in, const int* in_sizes, int n_in,
                              void* d_out, int out_size, void* d_ws, size_t ws_size,
                              hipStream_t stream) {
    const float* x     = (const float*)d_in[0];
    const int*   mask  = (const int*)  d_in[1];
    const float* Wq    = (const float*)d_in[2];
    const float* bq    = (const float*)d_in[3];
    const float* Wk    = (const float*)d_in[4];
    const float* bk    = (const float*)d_in[5];
    const float* Wv    = (const float*)d_in[6];
    const float* bv    = (const float*)d_in[7];
    const float* Wo    = (const float*)d_in[8];
    const float* bo    = (const float*)d_in[9];
    const float* ln1w  = (const float*)d_in[10];
    const float* ln1b  = (const float*)d_in[11];
    const float* W1    = (const float*)d_in[12];
    const float* b1    = (const float*)d_in[13];
    const float* W2    = (const float*)d_in[14];
    const float* b2    = (const float*)d_in[15];
    const float* ln2w  = (const float*)d_in[16];
    const float* ln2b  = (const float*)d_in[17];

    float* out_x = (float*)d_out;                                  // [B,T,D]
    float* attn  = (float*)d_out + (size_t)Bn * Tn * Dn;           // [B,H,T,T]

    // ---- workspace carve-up ----
    char* ws = (char*)d_ws;
    size_t off = 0;
    auto carve = [&](size_t bytes) { char* p = ws + off; off += (bytes + 255) & ~size_t(255); return p; };
    half_t* xn  = (half_t*)carve((size_t)Mrows * Dn * 2);          // ln1(x)
    half_t* wqp = (half_t*)carve((size_t)Dn * HD * 2);
    half_t* wkp = (half_t*)carve((size_t)Dn * HD * 2);
    half_t* wvp = (half_t*)carve((size_t)Dn * HD * 2);
    half_t* wop = (half_t*)carve((size_t)HD * Dn * 2);
    half_t* w1p = (half_t*)carve((size_t)Dn * DFFn * 2);
    half_t* w2p = (half_t*)carve((size_t)DFFn * Dn * 2);
    half_t* qh  = (half_t*)carve((size_t)Bn * Hn * Tn * Dn * 2);   // [B,H,T,D]
    half_t* kh  = (half_t*)carve((size_t)Bn * Hn * Tn * Dn * 2);   // [B,H,T,D]
    half_t* vt  = (half_t*)carve((size_t)Bn * Hn * Dn * Tn * 2);   // [B,H,D,T]
    half_t* ao  = (half_t*)carve((size_t)Mrows * HD * 2);          // concat heads
    float*  x1  = (float*) carve((size_t)Mrows * Dn * 4);          // attn sublayer out
    half_t* h2  = (half_t*)carve((size_t)Mrows * Dn * 2);          // ln2(x1)
    half_t* ff1 = (half_t*)carve((size_t)Mrows * DFFn * 2);        // relu(h@W1+b1)
    (void)ws_size; (void)n_in; (void)in_sizes; (void)out_size;

    // ---- 1. repack all weights to f16 B-fragment tiles ----
    pack_w_kernel<<<dim3(Dn / 32, HD / 16),  32, 0, stream>>>(Wq, wqp, Dn, HD);
    pack_w_kernel<<<dim3(Dn / 32, HD / 16),  32, 0, stream>>>(Wk, wkp, Dn, HD);
    pack_w_kernel<<<dim3(Dn / 32, HD / 16),  32, 0, stream>>>(Wv, wvp, Dn, HD);
    pack_w_kernel<<<dim3(HD / 32, Dn / 16),  32, 0, stream>>>(Wo, wop, HD, Dn);
    pack_w_kernel<<<dim3(Dn / 32, DFFn / 16),32, 0, stream>>>(W1, w1p, Dn, DFFn);
    pack_w_kernel<<<dim3(DFFn / 32, Dn / 16),32, 0, stream>>>(W2, w2p, DFFn, Dn);

    // ---- 2. LN1 ----
    layernorm_f16_kernel<<<Mrows, Dn, 0, stream>>>(x, ln1w, ln1b, xn);

    // ---- 3. Q/K/V projections (WMMA, 16x64 per wave) ----
    gemm_wmma_kernel<OUT_QK><<<dim3(HD / 64, Mrows / 16), 32, 0, stream>>>(
        xn, wqp, bq, nullptr, nullptr, qh, HD, Dn);
    gemm_wmma_kernel<OUT_QK><<<dim3(HD / 64, Mrows / 16), 32, 0, stream>>>(
        xn, wkp, bk, nullptr, nullptr, kh, HD, Dn);
    gemm_wmma_kernel<OUT_VT><<<dim3(HD / 64, Mrows / 16), 32, 0, stream>>>(
        xn, wvp, bv, nullptr, nullptr, vt, HD, Dn);

    // ---- 4. scores = QK^T/sqrt(D), mask (WMMA) ----
    attn_scores_kernel<<<dim3(Tn / 64, Tn / 16, Bn * Hn), 32, 0, stream>>>(
        qh, kh, mask, attn);

    // ---- 5. softmax rows (in place in d_out) ----
    softmax_rows_kernel<<<Bn * Hn * Tn, 256, 0, stream>>>(attn);

    // ---- 6. out = attn @ V (WMMA + async LDS staging, attn read once) ----
    attn_out_kernel<<<dim3(Tn / 16, Bn * Hn), 128, 0, stream>>>(attn, vt, ao);

    // ---- 7. output projection + residual (WMMA) ----
    gemm_wmma_kernel<OUT_F32RES><<<dim3(Dn / 64, Mrows / 16), 32, 0, stream>>>(
        ao, wop, bo, x, x1, nullptr, Dn, HD);

    // ---- 8. LN2 ----
    layernorm_f16_kernel<<<Mrows, Dn, 0, stream>>>(x1, ln2w, ln2b, h2);

    // ---- 9. FFN layer 1 + ReLU (WMMA) ----
    gemm_wmma_kernel<OUT_F16RELU><<<dim3(DFFn / 64, Mrows / 16), 32, 0, stream>>>(
        h2, w1p, b1, nullptr, nullptr, ff1, DFFn, Dn);

    // ---- 10. FFN layer 2 + residual -> d_out x region (WMMA) ----
    gemm_wmma_kernel<OUT_F32RES><<<dim3(Dn / 64, Mrows / 16), 32, 0, stream>>>(
        ff1, w2p, b2, x1, out_x, nullptr, Dn, DFFn);
}
